// LinearMultiheadAttention_24404004176365
// MI455X (gfx1250) — matrix operand order repaired and checked by reading, not compiled, not run on hardware
//
#include <hip/hip_runtime.h>
#include <cstdint>

// ---------------- problem constants ----------------
#define B_  4
#define N_  4096
#define C_  1024
#define H_  16
#define D_  64
#define EPS_ 1e-6f

typedef __bf16 bf16;
typedef __attribute__((ext_vector_type(16))) __bf16 v16bf;
typedef __attribute__((ext_vector_type(8)))  __bf16 v8bf;
typedef __attribute__((ext_vector_type(8)))  float  v8f;

#define WMMA_BF16(a, b, c) \
  __builtin_amdgcn_wmma_f32_16x16x32_bf16(false, (a), false, (b), (short)0, (c), false, false)

// ---------------------------------------------------------------------------
// Operand loaders (k order used here: contiguous, the same for A and B)
//
// A (16x32, 16-bit): lane l16 = row M; kb = 16*(lane>=16);
//   element t: K = kb + t
//   (caller passes p = rowBase + 16*half + kc)  -> two 16B loads
//
// B (32x16, 16-bit): lane l16 = col N; element t: K = 16*half + t
//   (caller passes p = colBase + 16*half + kc) -> two 16B loads
// ---------------------------------------------------------------------------
__device__ __forceinline__ v16bf lda_bf16(const bf16* __restrict__ p) {
  const v8bf* q = (const v8bf*)p;
  v8bf lo = q[0];     // K = kb + 0..7
  v8bf hi = q[1];     // K = kb + 8..15
  v16bf r;
#pragma unroll
  for (int t = 0; t < 8; ++t) { r[t] = lo[t]; r[t + 8] = hi[t]; }
  return r;
}
__device__ __forceinline__ v16bf ldb_bf16(const bf16* __restrict__ p) {
  const v8bf* q = (const v8bf*)p;
  v8bf lo = q[0];
  v8bf hi = q[1];
  v16bf r;
#pragma unroll
  for (int t = 0; t < 8; ++t) { r[t] = lo[t]; r[t + 8] = hi[t]; }
  return r;
}

// ---------------------------------------------------------------------------
// Kernel 0: streaming f32 -> bf16 conversion (x, W_qkv, W_out), 8 elems/thread
// ---------------------------------------------------------------------------
__global__ __launch_bounds__(256) void cvt_bf16_kernel(
    const float* __restrict__ in, bf16* __restrict__ out, int n) {
  const int i = (blockIdx.x * 256 + threadIdx.x) * 8;
  if (i >= n) return;
  const float4* p = (const float4*)(in + i);
  float4 a = p[0], b = p[1];
  v8bf o;
  o[0] = (bf16)a.x; o[1] = (bf16)a.y; o[2] = (bf16)a.z; o[3] = (bf16)a.w;
  o[4] = (bf16)b.x; o[5] = (bf16)b.y; o[6] = (bf16)b.z; o[7] = (bf16)b.w;
  *(v8bf*)(out + i) = o;
}

// ---------------------------------------------------------------------------
// Kernel 1: qkv = x @ W_qkv^T + b_qkv ; phi(); mask; scatter into
//   qf  [B,H,N,D]  (A-operand for stage 3)
//   kfT [B,H,D,N]  (A-operand for stage 2)
//   vT  [B,H,D,N]  (B-operand for stage 2)
// 32x32 macro-tile per wave (4 WMMAs per K-step); block = 4 waves = 64x64.
// `region` (q/k/v) is wave-uniform -> scalar branches in the epilogue.
// ---------------------------------------------------------------------------
__global__ __launch_bounds__(128) void qkv_phi_kernel(
    const bf16* __restrict__ xb, const bf16* __restrict__ Wqkv,
    const float* __restrict__ bqkv, const unsigned char* __restrict__ pad,
    bf16* __restrict__ qf, bf16* __restrict__ kfT, bf16* __restrict__ vT) {
  const int wave = threadIdx.x >> 5;
  const int lane = threadIdx.x & 31;
  const int l16 = lane & 15, half = lane >> 4;
  const int m0 = blockIdx.x * 64 + (wave >> 1) * 32;   // rows over B*N
  const int n0 = blockIdx.y * 64 + (wave & 1) * 32;    // cols over 3C

  const bf16* ap0 = xb   + (size_t)(m0 + l16)      * C_ + 16 * half;
  const bf16* ap1 = xb   + (size_t)(m0 + 16 + l16) * C_ + 16 * half;
  const bf16* bp0 = Wqkv + (size_t)(n0 + l16)      * C_ + 16 * half;
  const bf16* bp1 = Wqkv + (size_t)(n0 + 16 + l16) * C_ + 16 * half;

  v8f acc00 = {}, acc01 = {}, acc10 = {}, acc11 = {};
  for (int kc = 0; kc < C_; kc += 32) {
    v16bf a0 = lda_bf16(ap0 + kc);
    v16bf a1 = lda_bf16(ap1 + kc);
    v16bf b0 = ldb_bf16(bp0 + kc);
    v16bf b1 = ldb_bf16(bp1 + kc);
    acc00 = WMMA_BF16(a0, b0, acc00);
    acc01 = WMMA_BF16(a0, b1, acc01);
    acc10 = WMMA_BF16(a1, b0, acc10);
    acc11 = WMMA_BF16(a1, b1, acc11);
  }
  v8f accs[2][2] = {{acc00, acc01}, {acc10, acc11}};

  const int region = n0 >> 10;                 // wave-uniform: 0=q, 1=k, 2=v
  const int b_     = m0 >> 12;                 // wave-uniform batch index
  const size_t padBase = (size_t)b_ * N_;

#pragma unroll
  for (int mt = 0; mt < 2; ++mt) {
#pragma unroll
    for (int nt = 0; nt < 2; ++nt) {
      const v8f acc  = accs[mt][nt];
      const int colj = n0 + nt * 16 + l16;     // feature in [0, 3C)
      const float bias = bqkv[colj];
      const int j = colj & (C_ - 1);
      const int h = j >> 6, d = j & 63;
      const size_t bh = (size_t)b_ * H_ + h;
      const int nb = (m0 & (N_ - 1)) + mt * 16 + half * 8;
      if (region == 0) {
#pragma unroll
        for (int i = 0; i < 8; ++i) {
          float v = acc[i] + bias;
          float p = v > 0.f ? v + 1.f : __expf(v);         // elu(x)+1
          qf[(bh * N_ + nb + i) * D_ + d] = (bf16)p;
        }
      } else if (region == 1) {
#pragma unroll
        for (int i = 0; i < 8; ++i) {
          float v = acc[i] + bias;
          float p = v > 0.f ? v + 1.f : __expf(v);
          p = pad[padBase + nb + i] ? 0.f : p;             // True == PAD
          kfT[(bh * D_ + d) * N_ + nb + i] = (bf16)p;      // contiguous in n
        }
      } else {
#pragma unroll
        for (int i = 0; i < 8; ++i) {
          float v = acc[i] + bias;
          v = pad[padBase + nb + i] ? 0.f : v;
          vT[(bh * D_ + d) * N_ + nb + i] = (bf16)v;
        }
      }
    }
  }
}

// ---------------------------------------------------------------------------
// Kernel 2: per (b,h): kv[d,e] = sum_n kf[n,d] v[n,e]; z[d] = sum_n kf[n,d].
// 16 waves/block; wave owns one 16x16 tile of kv, streams K=N in 128 WMMAs.
// e0==0 waves compute z via an all-ones B operand (free on the matrix pipe).
// Stores kvT[e,d] bf16 (B-operand layout for stage 3) and z f32.
// ---------------------------------------------------------------------------
__global__ __launch_bounds__(512) void kv_state_kernel(
    const bf16* __restrict__ kfT, const bf16* __restrict__ vT,
    bf16* __restrict__ kvT, float* __restrict__ z) {
  const int bh   = blockIdx.x;
  const int wave = threadIdx.x >> 5;
  const int lane = threadIdx.x & 31;
  const int l16 = lane & 15, half = lane >> 4;
  const int d0 = (wave >> 2) * 16, e0 = (wave & 3) * 16;

  const bf16* ap = kfT + ((size_t)bh * D_ + d0 + l16) * N_ + 16 * half;
  const bf16* bp = vT  + ((size_t)bh * D_ + e0 + l16) * N_ + 16 * half;

  v16bf ones;
#pragma unroll
  for (int t = 0; t < 16; ++t) ones[t] = (bf16)1.0f;

  v8f acc = {}, accz = {};
  const bool doZ = (e0 == 0);              // wave-uniform -> EXEC stays all-1s
  for (int n = 0; n < N_; n += 32) {
    v16bf a = lda_bf16(ap + n);
    v16bf b = ldb_bf16(bp + n);
    acc = WMMA_BF16(a, b, acc);
    if (doZ) accz = WMMA_BF16(a, ones, accz);
  }

  bf16* op = kvT + ((size_t)bh * D_ + e0 + l16) * D_ + d0 + 8 * half;
#pragma unroll
  for (int i = 0; i < 8; ++i) op[i] = (bf16)acc[i];

  if (doZ && l16 == 0) {                   // z replicated over cols
    float* zp = z + bh * D_ + d0 + 8 * half;
#pragma unroll
    for (int i = 0; i < 8; ++i) zp[i] = accz[i];
  }
}

// ---------------------------------------------------------------------------
// Kernel 3: y_num = qf @ kv ; y_den = qf . z via lane-replicated z B-operand
// (result replicated across columns); y = num / max(den, eps).
// Wave computes a 16(n) x 64(e) strip: 8 num-WMMAs + 2 den-WMMAs.
// ---------------------------------------------------------------------------
__global__ __launch_bounds__(128) void y_kernel(
    const bf16* __restrict__ qf, const bf16* __restrict__ kvT,
    const float* __restrict__ z, bf16* __restrict__ y) {
  const int bh   = blockIdx.x;
  const int wave = threadIdx.x >> 5;
  const int n0   = blockIdx.y * 64 + wave * 16;
  const int lane = threadIdx.x & 31;
  const int l16 = lane & 15, half = lane >> 4;
  const int b_ = bh >> 4, h = bh & (H_ - 1);

  const bf16* ap = qf + ((size_t)bh * N_ + n0 + l16) * D_ + 16 * half;

  v8f acc[4] = {{}, {}, {}, {}};
  v8f accd = {};
#pragma unroll
  for (int kc = 0; kc < D_; kc += 32) {
    v16bf a = lda_bf16(ap + kc);
#pragma unroll
    for (int e = 0; e < 4; ++e) {
      const bf16* bp = kvT + ((size_t)bh * D_ + e * 16 + l16) * D_ + kc + 16 * half;
      acc[e] = WMMA_BF16(a, ldb_bf16(bp), acc[e]);
    }
    const float* zp = z + bh * D_ + kc + 16 * half;   // lane-replicated column
    v16bf zb;
#pragma unroll
    for (int t = 0; t < 16; ++t) zb[t] = (bf16)zp[t];
    accd = WMMA_BF16(a, zb, accd);
  }

#pragma unroll
  for (int i = 0; i < 8; ++i) {
    const int n = n0 + 8 * half + i;
    const float inv = 1.0f / fmaxf(accd[i], EPS_);
    const size_t row = ((size_t)b_ * N_ + n) * C_ + h * D_;
#pragma unroll
    for (int e = 0; e < 4; ++e)
      y[row + e * 16 + l16] = (bf16)(acc[e][i] * inv);
  }
}

// ---------------------------------------------------------------------------
// Kernel 4: out = y @ W_out^T + b_out (f32 out). 32x32 macro-tile per wave.
// ---------------------------------------------------------------------------
__global__ __launch_bounds__(128) void out_proj_kernel(
    const bf16* __restrict__ y, const bf16* __restrict__ Wout,
    const float* __restrict__ bout, float* __restrict__ out) {
  const int wave = threadIdx.x >> 5;
  const int lane = threadIdx.x & 31;
  const int l16 = lane & 15, half = lane >> 4;
  const int m0 = blockIdx.x * 64 + (wave >> 1) * 32;
  const int n0 = blockIdx.y * 64 + (wave & 1) * 32;

  const bf16* ap0 = y    + (size_t)(m0 + l16)      * C_ + 16 * half;
  const bf16* ap1 = y    + (size_t)(m0 + 16 + l16) * C_ + 16 * half;
  const bf16* bp0 = Wout + (size_t)(n0 + l16)      * C_ + 16 * half;
  const bf16* bp1 = Wout + (size_t)(n0 + 16 + l16) * C_ + 16 * half;

  v8f acc00 = {}, acc01 = {}, acc10 = {}, acc11 = {};
  for (int kc = 0; kc < C_; kc += 32) {
    v16bf a0 = lda_bf16(ap0 + kc);
    v16bf a1 = lda_bf16(ap1 + kc);
    v16bf b0 = ldb_bf16(bp0 + kc);
    v16bf b1 = ldb_bf16(bp1 + kc);
    acc00 = WMMA_BF16(a0, b0, acc00);
    acc01 = WMMA_BF16(a0, b1, acc01);
    acc10 = WMMA_BF16(a1, b0, acc10);
    acc11 = WMMA_BF16(a1, b1, acc11);
  }
  v8f accs[2][2] = {{acc00, acc01}, {acc10, acc11}};

#pragma unroll
  for (int mt = 0; mt < 2; ++mt) {
#pragma unroll
    for (int nt = 0; nt < 2; ++nt) {
      const float bias = bout[n0 + nt * 16 + l16];
      const int mrow = m0 + mt * 16 + 8 * half;
#pragma unroll
      for (int i = 0; i < 8; ++i)
        out[(size_t)(mrow + i) * C_ + n0 + nt * 16 + l16] = accs[mt][nt][i] + bias;
    }
  }
}

// ---------------------------------------------------------------------------
extern "C" void kernel_launch(void* const* d_in, const int* in_sizes, int n_in,
                              void* d_out, int out_size, void* d_ws, size_t ws_size,
                              hipStream_t stream) {
  const float* x    = (const float*)d_in[0];
  const float* Wqkv = (const float*)d_in[1];
  const float* bqkv = (const float*)d_in[2];
  const float* Wout = (const float*)d_in[3];
  const float* bout = (const float*)d_in[4];
  const unsigned char* pad = (const unsigned char*)d_in[5];  // jnp.bool_ (1B)
  float* out = (float*)d_out;

  // workspace carve-up (~169 MB)
  constexpr size_t SZ_QKV = (size_t)B_ * H_ * N_ * D_ * sizeof(bf16);  // 32 MB
  constexpr size_t SZ_KVT = (size_t)B_ * H_ * D_ * D_ * sizeof(bf16);
  constexpr size_t SZ_Z   = (size_t)B_ * H_ * D_ * sizeof(float);
  constexpr int   NX  = B_ * N_ * C_;       // x elements
  constexpr int   NWQ = 3 * C_ * C_;        // W_qkv elements
  constexpr int   NWO = C_ * C_;            // W_out elements
  char* ws = (char*)d_ws;
  size_t o = 0;
  bf16* qf   = (bf16*)(ws + o); o += SZ_QKV;
  bf16* kfT  = (bf16*)(ws + o); o += SZ_QKV;
  bf16* vT   = (bf16*)(ws + o); o += SZ_QKV;
  bf16* yb   = (bf16*)(ws + o); o += SZ_QKV;
  bf16* kvT  = (bf16*)(ws + o); o += SZ_KVT;
  float* z   = (float*)(ws + o); o += SZ_Z;
  bf16* xb   = (bf16*)(ws + o); o += (size_t)NX * sizeof(bf16);
  bf16* wqb  = (bf16*)(ws + o); o += (size_t)NWQ * sizeof(bf16);
  bf16* wob  = (bf16*)(ws + o); o += (size_t)NWO * sizeof(bf16);
  (void)in_sizes; (void)n_in; (void)out_size; (void)ws_size;

  // 0) one-time f32 -> bf16 operand conversion
  cvt_bf16_kernel<<<NX  / (256 * 8), 256, 0, stream>>>(x,    xb,  NX);
  cvt_bf16_kernel<<<NWQ / (256 * 8), 256, 0, stream>>>(Wqkv, wqb, NWQ);
  cvt_bf16_kernel<<<NWO / (256 * 8), 256, 0, stream>>>(Wout, wob, NWO);

  // 1) QKV projection + phi + mask (M=16384, Nout=3072, K=1024)
  qkv_phi_kernel<<<dim3((B_ * N_) / 64, (3 * C_) / 64), 128, 0, stream>>>(
      xb, wqb, bqkv, pad, qf, kfT, vT);

  // 2) kv state + z per (b,h)
  kv_state_kernel<<<dim3(B_ * H_), 512, 0, stream>>>(kfT, vT, kvT, z);

  // 3) y = (qf @ kv) / max(qf . z, eps)
  y_kernel<<<dim3(B_ * H_, N_ / 64), 128, 0, stream>>>(qf, kvT, z, yb);

  // 4) output projection (M=16384, Nout=1024, K=1024)
  out_proj_kernel<<<dim3((B_ * N_) / 64, C_ / 64), 128, 0, stream>>>(
      yb, wob, bout, out);
}
